// CrossAttention_17781164605499
// MI455X (gfx1250) — matrix operand order repaired and checked by reading, not compiled, not run on hardware
//
#include <hip/hip_runtime.h>
#include <hip/hip_bf16.h>

// ---------------------------------------------------------------------------
// CDNA5 (gfx1250) cross-attention.
// All matrices pre-packed into WMMA fragment-major bf16 chunks:
//   one chunk = one 16x32 (A) or 32x16 (B) fragment = 32 lanes x 16 bf16 = 1KB,
//   so every fragment load is a single 32B vector load (2 x b128).
// GEMM loops are fully unrolled with per-k-step fragment arrays so loads and
// v_wmma_f32_16x16x32_bf16 can overlap (non-zero s_wait_loadcnt).
// ---------------------------------------------------------------------------

typedef __bf16 bf16_t;
typedef __attribute__((ext_vector_type(16))) __bf16 v16bf;
typedef __attribute__((ext_vector_type(8)))  float  v8f;

#define B_    16
#define NQ_   4096
#define NK_   77
#define NKP   80      // keys padded for QK^T (5 x 16 n-tiles)
#define NKP2  96      // keys padded for PV   (3 x 32 k-steps)
#define QD    512
#define CD    768
#define H_    8
#define DH    64
#define INNER 512
#define SCALE_ 0.125f // 64^-0.5

// ---- helpers ---------------------------------------------------------------

__device__ __forceinline__ unsigned short bf16_bits(float f) {
  union { __hip_bfloat16 h; unsigned short u; } cv;
  cv.h = __float2bfloat16(f);
  return cv.u;
}
__device__ __forceinline__ unsigned int pk_bf16(float lo, float hi) {
  float2 f; f.x = lo; f.y = hi;
  union { __hip_bfloat162 h; unsigned int u; } cv;
  cv.h = __float22bfloat162_rn(f);              // -> v_cvt_pk_bf16_f32
  return cv.u;
}
__device__ __forceinline__ v8f wmma_bf16(v16bf a, v16bf b, v8f c) {
  return __builtin_amdgcn_wmma_f32_16x16x32_bf16(
      false, a, false, b, (short)0, c, false, false);
}
// assemble a fragment from two 16B runs (used by LDS bounce reads)
__device__ __forceinline__ v16bf frag_from2(const unsigned short* p0,
                                            const unsigned short* p1) {
  union { uint4 q[2]; v16bf v; } cv;
  cv.q[0] = *(const uint4*)p0;
  cv.q[1] = *(const uint4*)p1;
  return cv.v;
}

// Fragment-major chunk semantics (per ISA VGPR layouts):
//  A chunk (16x32):  lane = row(lane&15), kb = 8*(lane>=16);
//                    j<8 -> k=kb+j, j>=8 -> k=kb+16+(j-8)
//  B chunk (32x16):  lane = col(lane&15) + 16*khalf; k = 8*khalf + j (j<8), 8*khalf + 16 + (j-8) (j>=8)

// ---- pack x (B*NQ x 512 f32) into A-frag chunks [mtile][kstep][512] --------

__global__ void __launch_bounds__(256)
xpack_kernel(const float* __restrict__ x, unsigned short* __restrict__ xp) {
  const int lane = threadIdx.x & 31, wave = threadIdx.x >> 5;
  const int g = blockIdx.x * 8 + wave;          // 4096*16 tasks
  const int mtile = g >> 4, ks = g & 15;
  const int row = mtile * 16 + (lane & 15);
  const int kb = (lane >> 4) << 3;
  const float* s = x + (size_t)row * QD + ks * 32 + kb;
  float4 f0 = ((const float4*)s)[0];
  float4 f1 = ((const float4*)s)[1];
  float4 f2 = ((const float4*)(s + 16))[0];
  float4 f3 = ((const float4*)(s + 16))[1];
  uint4 u0, u1;
  u0.x = pk_bf16(f0.x, f0.y); u0.y = pk_bf16(f0.z, f0.w);
  u0.z = pk_bf16(f1.x, f1.y); u0.w = pk_bf16(f1.z, f1.w);
  u1.x = pk_bf16(f2.x, f2.y); u1.y = pk_bf16(f2.z, f2.w);
  u1.z = pk_bf16(f3.x, f3.y); u1.w = pk_bf16(f3.z, f3.w);
  uint4* d = (uint4*)(xp + (size_t)g * 512 + lane * 16);
  d[0] = u0; d[1] = u1;
}

// ---- pack a (ksteps*32 x 512 f32) weight into B-frag chunks [ks][ntile] ----

__global__ void __launch_bounds__(256)
wpack_kernel(const float* __restrict__ W, unsigned short* __restrict__ wp) {
  const int lane = threadIdx.x & 31, wave = threadIdx.x >> 5;
  const int g = blockIdx.x * 8 + wave;          // ksteps*32 tasks
  const int ks = g >> 5, nt = g & 31;
  const int col = nt * 16 + (lane & 15);
  const int khalf = lane >> 4;
  const float* s = W + ((size_t)ks * 32 + khalf * 8) * INNER + col;
  float v[16];
#pragma unroll
  for (int j = 0; j < 16; ++j) v[j] = s[(size_t)(j + (j & 8)) * INNER];
  uint4 u0, u1;
  u0.x = pk_bf16(v[0], v[1]);   u0.y = pk_bf16(v[2], v[3]);
  u0.z = pk_bf16(v[4], v[5]);   u0.w = pk_bf16(v[6], v[7]);
  u1.x = pk_bf16(v[8], v[9]);   u1.y = pk_bf16(v[10], v[11]);
  u1.z = pk_bf16(v[12], v[13]); u1.w = pk_bf16(v[14], v[15]);
  uint4* d = (uint4*)(wp + (size_t)g * 512 + lane * 16);
  d[0] = u0; d[1] = u1;
}

// ---- pack context (16 x 77 x 768) into A-frag chunks, rows>=77 zeroed ------

__global__ void __launch_bounds__(256)
cpack_kernel(const float* __restrict__ ctx, unsigned short* __restrict__ cp) {
  const int lane = threadIdx.x & 31, wave = threadIdx.x >> 5;
  const int g = blockIdx.x * 8 + wave;          // 16*6*24 = 2304 tasks
  const int b = g / (6 * 24);
  const int rem = g % (6 * 24);
  const int mt = rem / 24, ks = rem % 24;
  const int key = mt * 16 + (lane & 15);
  const int kb = (lane >> 4) << 3;
  uint4 u0 = {0, 0, 0, 0}, u1 = {0, 0, 0, 0};
  if (key < NK_) {
    const float* s = ctx + ((size_t)b * NK_ + key) * CD + ks * 32 + kb;
    float4 f0 = ((const float4*)s)[0];
    float4 f1 = ((const float4*)s)[1];
    float4 f2 = ((const float4*)(s + 16))[0];
    float4 f3 = ((const float4*)(s + 16))[1];
    u0.x = pk_bf16(f0.x, f0.y); u0.y = pk_bf16(f0.z, f0.w);
    u0.z = pk_bf16(f1.x, f1.y); u0.w = pk_bf16(f1.z, f1.w);
    u1.x = pk_bf16(f2.x, f2.y); u1.y = pk_bf16(f2.z, f2.w);
    u1.z = pk_bf16(f3.x, f3.y); u1.w = pk_bf16(f3.z, f3.w);
  }
  uint4* d = (uint4*)(cp + (size_t)g * 512 + lane * 16);
  d[0] = u0; d[1] = u1;
}

// ---- K/V projection: packed frags in, K^T-frag / V-frag chunks out ---------

__global__ void __launch_bounds__(192)
kv_kernel(const unsigned short* __restrict__ cp, const unsigned short* __restrict__ wkp,
          const unsigned short* __restrict__ wvp, unsigned short* __restrict__ ktp,
          unsigned short* __restrict__ vp) {
  __shared__ __align__(32) unsigned short Dld[NKP2 * DH];  // 12 KB bounce
  const int h = blockIdx.x, which = blockIdx.y, b = blockIdx.z;
  const int lane = threadIdx.x & 31, wave = threadIdx.x >> 5;   // 6 waves
  const unsigned short* wp = which ? wvp : wkp;
  v8f zero = {};
  v8f acc[4] = {zero, zero, zero, zero};
  const unsigned short* ap = cp + (((size_t)b * 6 + wave) * 24) * 512 + lane * 16;
  const unsigned short* bp = wp + (size_t)h * 4 * 512 + lane * 16;
#pragma unroll
  for (int ks = 0; ks < 24; ++ks) {
    v16bf a = *(const v16bf*)(ap + (size_t)ks * 512);
    v16bf bfr[4];
#pragma unroll
    for (int nt = 0; nt < 4; ++nt)
      bfr[nt] = *(const v16bf*)(bp + ((size_t)ks * 32 + nt) * 512);
#pragma unroll
    for (int nt = 0; nt < 4; ++nt)
      acc[nt] = wmma_bf16(a, bfr[nt], acc[nt]);
  }
  const int col = lane & 15, rsh = (lane >> 4) << 3;
  const int m0 = wave * 16;                     // keys; rows >=77 are exact 0
  if (which == 0) {                             // K: Dld[key][d]
#pragma unroll
    for (int nt = 0; nt < 4; ++nt)
#pragma unroll
      for (int r = 0; r < 8; ++r)
        Dld[(size_t)(m0 + rsh + r) * DH + nt * 16 + col] = bf16_bits(acc[nt][r]);
  } else {                                      // V: Dld[d][key], packed stores
#pragma unroll
    for (int nt = 0; nt < 4; ++nt) {
      unsigned int* dd = (unsigned int*)&Dld[(size_t)(nt * 16 + col) * NKP2 + m0 + rsh];
#pragma unroll
      for (int r = 0; r < 4; ++r)
        dd[r] = pk_bf16(acc[nt][2 * r], acc[nt][2 * r + 1]);
    }
  }
  __syncthreads();
  const size_t bh = (size_t)b * H_ + h;
  const int khalf16 = (lane >> 4) << 3;
  if (which == 0) {                             // emit K^T B-frag chunks (2x5)
    for (int c = wave; c < 10; c += 6) {
      const int kt = c / 5, nt5 = c % 5;
      const unsigned short* p = &Dld[(size_t)(nt5 * 16 + col) * DH + kt * 32 + khalf16];
      v16bf v = frag_from2(p, p + 16);
      *(v16bf*)(ktp + (bh * 10 + c) * 512 + lane * 16) = v;
    }
  } else {                                      // emit V B-frag chunks (3x4)
    for (int c = wave; c < 12; c += 6) {
      const int kt = c / 4, nt = c % 4;
      const unsigned short* p = &Dld[(size_t)(nt * 16 + col) * NKP2 + kt * 32 + khalf16];
      v16bf v = frag_from2(p, p + 16);
      *(v16bf*)(vp + (bh * 12 + c) * 512 + lane * 16) = v;
    }
  }
}

// ---- Q projection: LDS-free WMMA GEMM, output re-packed to A-frag chunks ---

__global__ void __launch_bounds__(256)
qproj_kernel(const unsigned short* __restrict__ xp, const unsigned short* __restrict__ wqp,
             unsigned short* __restrict__ qp) {
  __shared__ __align__(32) unsigned short Qb[8][16][128];   // 32 KB bounce
  const int lane = threadIdx.x & 31, wave = threadIdx.x >> 5;
  const int g = blockIdx.x * 8 + wave;          // 4096 mtiles x 4 col-groups
  const int mtile = g >> 2, cg = g & 3;
  v8f zero = {};
  v8f acc[8] = {zero, zero, zero, zero, zero, zero, zero, zero};
  const unsigned short* ap = xp + (size_t)mtile * 16 * 512 + lane * 16;
  const unsigned short* bp = wqp + (size_t)cg * 8 * 512 + lane * 16;
#pragma unroll
  for (int ks = 0; ks < 16; ++ks) {
    v16bf a = *(const v16bf*)(ap + (size_t)ks * 512);
    v16bf bfr[8];
#pragma unroll
    for (int nt = 0; nt < 8; ++nt)
      bfr[nt] = *(const v16bf*)(bp + ((size_t)ks * 32 + nt) * 512);
#pragma unroll
    for (int nt = 0; nt < 8; ++nt)
      acc[nt] = wmma_bf16(a, bfr[nt], acc[nt]);
  }
  const int col = lane & 15, rsh = (lane >> 4) << 3;
#pragma unroll
  for (int nt = 0; nt < 8; ++nt)
#pragma unroll
    for (int r = 0; r < 8; ++r)
      Qb[wave][rsh + r][nt * 16 + col] = bf16_bits(acc[nt][r]);
  __syncthreads();
  const int bi = mtile >> 8, qt = mtile & 255;  // batch, q-tile
  const int row = lane & 15, kb = (lane >> 4) << 3;
#pragma unroll
  for (int db = 0; db < 4; ++db) {              // 4 x 32-wide d-blocks
    const int gn0 = cg * 128 + db * 32;
    const int h = gn0 >> 6, ks2 = (gn0 >> 5) & 1;
    const unsigned short* p = &Qb[wave][row][db * 32 + kb];
    v16bf v = frag_from2(p, p + 16);
    *(v16bf*)(qp + (((size_t)(bi * H_ + h) * 256 + qt) * 2 + ks2) * 512 + lane * 16) = v;
  }
}

// ---- attention: all fragments preloaded as single 32B loads ----------------

__global__ void __launch_bounds__(128)
attn_kernel(const unsigned short* __restrict__ qp, const unsigned short* __restrict__ ktp,
            const unsigned short* __restrict__ vp, unsigned short* __restrict__ op) {
  __shared__ __align__(32) float          Sld[4][16][NKP];       // 20 KB
  __shared__ __align__(32) unsigned short Pf[4][3][32][16];      // 12 KB (A-frag)
  __shared__ __align__(32) unsigned short Ob[4][16][64];         //  8 KB bounce
  const int lane = threadIdx.x & 31, wave = threadIdx.x >> 5;
  const int h = blockIdx.y, b = blockIdx.z;
  const int qt = blockIdx.x * 4 + wave;
  const size_t bh = (size_t)b * H_ + h;
  v8f zero = {};
  v8f accS[5] = {zero, zero, zero, zero, zero};

  // S = Q K^T: preload 2 Q A-frags + all 10 K^T B-frags (contiguous chunks)
  v16bf aq[2];
#pragma unroll
  for (int kt = 0; kt < 2; ++kt)
    aq[kt] = *(const v16bf*)(qp + ((bh * 256 + qt) * 2 + kt) * 512 + lane * 16);
  v16bf kf[10];
#pragma unroll
  for (int c = 0; c < 10; ++c)
    kf[c] = *(const v16bf*)(ktp + (bh * 10 + c) * 512 + lane * 16);
#pragma unroll
  for (int kt = 0; kt < 2; ++kt)
#pragma unroll
    for (int nt = 0; nt < 5; ++nt)
      accS[nt] = wmma_bf16(aq[kt], kf[kt * 5 + nt], accS[nt]);

  const int col = lane & 15, rsh = (lane >> 4) << 3;
#pragma unroll
  for (int nt = 0; nt < 5; ++nt)
#pragma unroll
    for (int r = 0; r < 8; ++r)
      Sld[wave][rsh + r][nt * 16 + col] = accS[nt][r] * SCALE_;
  __syncthreads();
  if (lane < 16) {                              // row softmax over 77 keys
    float* Srow = &Sld[wave][lane][0];
    float mx = -3.0e38f;
    for (int j = 0; j < NK_; ++j) mx = fmaxf(mx, Srow[j]);
    float sum = 0.f;
    for (int j = 0; j < NK_; ++j) { float e = __expf(Srow[j] - mx); Srow[j] = e; sum += e; }
    const float inv = 1.0f / sum;
    for (int kk = 0; kk < NKP2; ++kk) {         // scatter into A-frag layout
      const int kt = kk >> 5, rem = kk & 31, sub = rem >> 3;
      const int jj = (rem & 7) + ((sub >> 1) << 3);
      const int lt = lane + ((sub & 1) << 4);
      Pf[wave][kt][lt][jj] = (kk < NK_) ? bf16_bits(Srow[kk] * inv) : (unsigned short)0;
    }
  }
  __syncthreads();

  // O = P V: preload all 12 V B-frags (contiguous chunks)
  v16bf vf[12];
#pragma unroll
  for (int c = 0; c < 12; ++c)
    vf[c] = *(const v16bf*)(vp + (bh * 12 + c) * 512 + lane * 16);
  v8f accO[4] = {zero, zero, zero, zero};
#pragma unroll
  for (int kt = 0; kt < 3; ++kt) {
    v16bf a = *(const v16bf*)&Pf[wave][kt][lane][0];
#pragma unroll
    for (int nt = 0; nt < 4; ++nt)
      accO[nt] = wmma_bf16(a, vf[kt * 4 + nt], accO[nt]);
  }
#pragma unroll
  for (int nt = 0; nt < 4; ++nt)
#pragma unroll
    for (int r = 0; r < 8; ++r)
      Ob[wave][rsh + r][nt * 16 + col] = bf16_bits(accO[nt][r]);
  __syncthreads();
  const size_t mtg = (size_t)b * 256 + qt;      // global m-tile of O
  const int row = lane & 15, kb = (lane >> 4) << 3;
#pragma unroll
  for (int ksl = 0; ksl < 2; ++ksl) {           // emit oproj A-frag chunks
    const unsigned short* p = &Ob[wave][row][ksl * 32 + kb];
    v16bf v = frag_from2(p, p + 16);
    *(v16bf*)(op + (mtg * 16 + h * 2 + ksl) * 512 + lane * 16) = v;
  }
}

// ---- output projection: LDS-free WMMA GEMM + bias, f32 out -----------------

__global__ void __launch_bounds__(256)
oproj_kernel(const unsigned short* __restrict__ op, const unsigned short* __restrict__ wop,
             const float* __restrict__ bo, float* __restrict__ out) {
  const int lane = threadIdx.x & 31, wave = threadIdx.x >> 5;
  const int g = blockIdx.x * 8 + wave;
  const int mtile = g >> 2, cg = g & 3;
  v8f zero = {};
  v8f acc[8] = {zero, zero, zero, zero, zero, zero, zero, zero};
  const unsigned short* ap = op + (size_t)mtile * 16 * 512 + lane * 16;
  const unsigned short* bp = wop + (size_t)cg * 8 * 512 + lane * 16;
#pragma unroll
  for (int ks = 0; ks < 16; ++ks) {
    v16bf a = *(const v16bf*)(ap + (size_t)ks * 512);
    v16bf bfr[8];
#pragma unroll
    for (int nt = 0; nt < 8; ++nt)
      bfr[nt] = *(const v16bf*)(bp + ((size_t)ks * 32 + nt) * 512);
#pragma unroll
    for (int nt = 0; nt < 8; ++nt)
      acc[nt] = wmma_bf16(a, bfr[nt], acc[nt]);
  }
  const int col = lane & 15, rsh = (lane >> 4) << 3;
#pragma unroll
  for (int nt = 0; nt < 8; ++nt) {
    const int gn = cg * 128 + nt * 16 + col;
    const float bias = bo[gn];
#pragma unroll
    for (int r = 0; r < 8; ++r) {
      const int gm = mtile * 16 + rsh + r;
      out[(size_t)gm * QD + gn] = acc[nt][r] + bias;
    }
  }
}

// ---- launch ---------------------------------------------------------------

extern "C" void kernel_launch(void* const* d_in, const int* in_sizes, int n_in,
                              void* d_out, int out_size, void* d_ws, size_t ws_size,
                              hipStream_t stream) {
  const float* x   = (const float*)d_in[0];   // (16, 4096, 512)
  const float* ctx = (const float*)d_in[1];   // (16, 77, 768)
  const float* Wq  = (const float*)d_in[2];   // (512, 512)
  const float* Wk  = (const float*)d_in[3];   // (768, 512)
  const float* Wv  = (const float*)d_in[4];   // (768, 512)
  const float* Wo  = (const float*)d_in[5];   // (512, 512)
  const float* bo  = (const float*)d_in[6];   // (512,)
  float* out = (float*)d_out;                 // (16, 4096, 512)

  unsigned short* xpack = (unsigned short*)d_ws;           // 64 MB
  unsigned short* opack = xpack;                           // aliases xpack (dead after qproj)
  unsigned short* qpack = xpack + (size_t)33554432;        // 64 MB
  unsigned short* wqp   = qpack + (size_t)33554432;        // 512 KB
  unsigned short* wop   = wqp   + (size_t)262144;          // 512 KB
  unsigned short* wkp   = wop   + (size_t)262144;          // 768 KB
  unsigned short* wvp   = wkp   + (size_t)393216;          // 768 KB
  unsigned short* cpk   = wvp   + (size_t)393216;          // 2.25 MB
  unsigned short* ktp   = cpk   + (size_t)1179648;         // 1.25 MB
  unsigned short* vpk   = ktp   + (size_t)655360;          // 1.5 MB

  xpack_kernel<<<8192, 256, 0, stream>>>(x, xpack);
  wpack_kernel<<<64, 256, 0, stream>>>(Wq, wqp);
  wpack_kernel<<<64, 256, 0, stream>>>(Wo, wop);
  wpack_kernel<<<96, 256, 0, stream>>>(Wk, wkp);
  wpack_kernel<<<96, 256, 0, stream>>>(Wv, wvp);
  cpack_kernel<<<288, 256, 0, stream>>>(ctx, cpk);
  kv_kernel<<<dim3(H_, 2, B_), 192, 0, stream>>>(cpk, wkp, wvp, ktp, vpk);
  qproj_kernel<<<2048, 256, 0, stream>>>(xpack, wqp, qpack);
  attn_kernel<<<dim3(NQ_ / 64, H_, B_), 128, 0, stream>>>(qpack, ktp, vpk, opack);
  oproj_kernel<<<2048, 256, 0, stream>>>(opack, wop, bo, out);
}
